// HierarchicalMemorySystem_79164837200280
// MI455X (gfx1250) — matrix operand order repaired and checked by reading, not compiled, not run on hardware
//
#include <hip/hip_runtime.h>

// ---------------------------------------------------------------------------
// Types for CDNA5 WMMA
// ---------------------------------------------------------------------------
typedef __attribute__((ext_vector_type(16))) _Float16 v16h;
typedef __attribute__((ext_vector_type(8)))  _Float16 v8h;
typedef __attribute__((ext_vector_type(4)))  _Float16 v4h;
typedef __attribute__((ext_vector_type(2)))  _Float16 v2h;
typedef __attribute__((ext_vector_type(8)))  float    v8f;

union V16H { v16h v; v8h h8[2]; _Float16 h[16]; };

// activation codes
#define ACT_NONE    0
#define ACT_ELU1    1   // elu(x)+1
#define ACT_SIGMOID 2
#define ACT_SILU    3

#define TM 64       // C rows per block (4 waves x 16)
#define TN 32       // C cols per block (2 wmma N-tiles per wave)
#define TK 32       // K step (16x16x32 f16 WMMA)
#define APITCH 40   // LDS row pitch in halfs (80B = 16B multiple)
#define BPITCH 40

__device__ __forceinline__ float apply_act(float x, int act)
{
    if      (act == ACT_ELU1)    return (x > 0.0f) ? (x + 1.0f) : __expf(x);
    else if (act == ACT_SIGMOID) return 1.0f / (1.0f + __expf(-x));
    else if (act == ACT_SILU)    return x / (1.0f + __expf(-x));
    return x;
}

// ---------------------------------------------------------------------------
// Generic tiled WMMA GEMM:  C[z] = act( A[z] @ B[z] (+bias) ) * scale
// A: (Mrows x K) row-major, lda.   B: NN (K x N, ldb) or NT (N x K, ldb).
// z = blockIdx.z decomposed as (zb = z/Hdiv, zh = z%Hdiv), element offsets
// zb*s?b + zh*s?h applied to A/B/C base pointers.
// Requires: K % 32 == 0, Ncols % 32 == 0 (no N guards!), lda % 4 == 0,
// A/C base 16B-aligned. Row guard only taken when Mrows % 64 != 0.
// ---------------------------------------------------------------------------
__global__ __launch_bounds__(128) void gemm_wmma(
    const float* __restrict__ A, const float* __restrict__ B,
    float* __restrict__ C, const float* __restrict__ bias,
    int Mrows, int Ncols, int K,
    int lda, int ldb, int ldc,
    int Hdiv,
    long sAb, long sAh, long sBb, long sBh, long sCb, long sCh,
    int transB, int act, float scale)
{
    int zb = blockIdx.z / Hdiv, zh = blockIdx.z % Hdiv;
    A += zb * sAb + zh * sAh;
    B += zb * sBb + zh * sBh;
    C += zb * sCb + zh * sCh;

    const int tileM = blockIdx.x * TM;
    const int tileN = blockIdx.y * TN;
    const int tid   = threadIdx.x;
    const int wave  = tid >> 5;     // 0..3
    const int lane  = tid & 31;

    __shared__ __align__(16) _Float16 As[TM][APITCH];   // As[r][k]
    __shared__ __align__(16) _Float16 BsT[TN][BPITCH];  // BsT[n][k]  (transposed)

    v8f acc0 = {}, acc1 = {};
    const bool fullM = (tileM + TM) <= Mrows;

    // per-thread staging coordinates (loop-invariant)
    const int ar  = tid >> 3;          // A row slot (0..15 step: +16 per i... see below)
    const int ac4 = tid & 7;           // A float4 column slot
    const int bn  = tid & 31;          // B n slot
    const int bk2 = tid >> 5;          // B k-pair slot base (0..3, +4 per i)

    for (int k0 = 0; k0 < K; k0 += TK) {
        // ================= stage A tile: 64 x 32 f32 =================
        float4 ta[4];
        if (fullM) {   // uniform branch: no exec masking, loads batch freely
            #pragma unroll
            for (int i = 0; i < 4; ++i) {
                int r = ar + i * 16;
                ta[i] = *(const float4*)(A + (long)(tileM + r) * lda + (k0 + ac4 * 4));
            }
            if (k0 + TK < K)
                __builtin_prefetch(A + (long)(tileM + ar) * lda + (k0 + TK), 0, 3);
        } else {
            #pragma unroll
            for (int i = 0; i < 4; ++i) {
                int r  = ar + i * 16;
                int gr = tileM + r;
                ta[i] = (gr < Mrows) ? *(const float4*)(A + (long)gr * lda + (k0 + ac4 * 4))
                                     : make_float4(0.f, 0.f, 0.f, 0.f);
            }
        }
        #pragma unroll
        for (int i = 0; i < 4; ++i) {
            int r = ar + i * 16;
            v4h h4;
            h4[0] = (_Float16)ta[i].x; h4[1] = (_Float16)ta[i].y;
            h4[2] = (_Float16)ta[i].z; h4[3] = (_Float16)ta[i].w;
            *(v4h*)&As[r][ac4 * 4] = h4;             // ds_store_b64
        }

        // ====== stage B tile transposed: 32(K) x 32(N), K-pairs ======
        float tb0[4], tb1[4];
        #pragma unroll
        for (int i = 0; i < 4; ++i) {
            int kk = k0 + (bk2 + i * 4) * 2;
            int gn = tileN + bn;                     // always < Ncols (N%32==0)
            if (transB) {
                tb0[i] = B[(long)gn * ldb + kk];
                tb1[i] = B[(long)gn * ldb + kk + 1];
            } else {
                tb0[i] = B[(long)kk * ldb + gn];
                tb1[i] = B[(long)(kk + 1) * ldb + gn];
            }
        }
        #pragma unroll
        for (int i = 0; i < 4; ++i) {
            v2h p;
            p[0] = (_Float16)tb0[i];
            p[1] = (_Float16)tb1[i];
            *(v2h*)&BsT[bn][(bk2 + i * 4) * 2] = p;  // ds_store_b32
        }
        __syncthreads();

        // ---- gather fragments (vectorized ds_load_b128) ----
        const int g  = lane >> 4;            // half-wave group
        const int nn = lane & 15;
        const int mr = nn + wave * 16;
        V16H a, b0, b1;
        a.h8[0]  = *(const v8h*)&As[mr][g * 16];       // K = g*16 + 0..7
        a.h8[1]  = *(const v8h*)&As[mr][g * 16 + 8];   // K = g*16 + 8..15
        b0.h8[0] = *(const v8h*)&BsT[nn][g * 16];      // K = g*16 + 0..7
        b0.h8[1] = *(const v8h*)&BsT[nn][g * 16 + 8];  // K = g*16 + 8..15
        b1.h8[0] = *(const v8h*)&BsT[16 + nn][g * 16];
        b1.h8[1] = *(const v8h*)&BsT[16 + nn][g * 16 + 8];

        acc0 = __builtin_amdgcn_wmma_f32_16x16x32_f16(
                   false, a.v, false, b0.v, (short)0, acc0, false, false);
        acc1 = __builtin_amdgcn_wmma_f32_16x16x32_f16(
                   false, a.v, false, b1.v, (short)0, acc1, false, false);
        __syncthreads();
    }

    // ---- epilogue: lane (g,n) owns rows M = wave*16 + g*8 + r ----
    const int g  = lane >> 4;
    const int n  = lane & 15;
    const int col0 = tileN + n;
    const int col1 = tileN + 16 + n;
    const float bv0 = bias ? bias[col0] : 0.0f;
    const float bv1 = bias ? bias[col1] : 0.0f;
    if (fullM) {
        #pragma unroll
        for (int r = 0; r < 8; ++r) {
            int row = tileM + wave * 16 + g * 8 + r;
            C[(long)row * ldc + col0] = apply_act(acc0[r] + bv0, act) * scale;
            C[(long)row * ldc + col1] = apply_act(acc1[r] + bv1, act) * scale;
        }
    } else {
        #pragma unroll
        for (int r = 0; r < 8; ++r) {
            int row = tileM + wave * 16 + g * 8 + r;
            if (row < Mrows) {
                C[(long)row * ldc + col0] = apply_act(acc0[r] + bv0, act) * scale;
                C[(long)row * ldc + col1] = apply_act(acc1[r] + bv1, act) * scale;
            }
        }
    }
}

// ---------------------------------------------------------------------------
// WorkingMemory gated-linear-attention scan (recurrence form):
//   S_t[m,d] = f_t[m] S_{t-1}[m,d] + k_t[m] v_t[d]
//   z_t[m]   = f_t[m] z_{t-1}[m]   + k_t[m]
//   y_t[d]   = (sum_m q_t[m] S_t[m,d]) / (q_t . z_t + 1e-6)
// One thread per d (64 d's per block), state column S[64] in VGPRs.
// ---------------------------------------------------------------------------
__global__ __launch_bounds__(64) void wm_scan(
    const float* __restrict__ q, const float* __restrict__ k,
    const float* __restrict__ f, const float* __restrict__ v,
    float* __restrict__ y, int L)
{
    const int b   = blockIdx.y;
    const int tid = threadIdx.x;
    const int d   = blockIdx.x * 64 + tid;

    __shared__ float sq[64], sk[64], sf[64], red[64];

    float S[64];
    #pragma unroll
    for (int m = 0; m < 64; ++m) S[m] = 0.0f;
    float z = 0.0f;

    const float* qb = q + (long)b * L * 64;
    const float* kb = k + (long)b * L * 64;
    const float* fb = f + (long)b * L * 64;
    const float* vb = v + (long)b * L * 512;
    float*       yb = y + (long)b * L * 512;

    for (int t = 0; t < L; ++t) {
        sq[tid] = qb[t * 64 + tid];
        sk[tid] = kb[t * 64 + tid];
        sf[tid] = fb[t * 64 + tid];
        __syncthreads();

        z = sf[tid] * z + sk[tid];
        red[tid] = sq[tid] * z;
        __syncthreads();
        for (int s = 32; s > 0; s >>= 1) {
            if (tid < s) red[tid] += red[tid + s];
            __syncthreads();
        }
        const float den = red[0] + 1e-6f;

        const float vv = vb[t * 512 + d];
        float num = 0.0f;
        #pragma unroll
        for (int m = 0; m < 64; ++m) {
            float sm = sf[m] * S[m] + sk[m] * vv;
            S[m] = sm;
            num += sq[m] * sm;
        }
        yb[t * 512 + d] = num / den;
        __syncthreads();
    }
}

// ---------------------------------------------------------------------------
// In-place LayerNorm over last dim (512), one block (128 thr) per row.
// ---------------------------------------------------------------------------
__global__ __launch_bounds__(128) void layernorm_ip(
    float* __restrict__ y, const float* __restrict__ g,
    const float* __restrict__ b, int Dm)
{
    const int row = blockIdx.x;
    const int tid = threadIdx.x;
    float* p = y + (long)row * Dm;

    float vals[4];
    float s = 0.0f, s2 = 0.0f;
    #pragma unroll
    for (int i = 0; i < 4; ++i) {
        float x = p[tid + i * 128];
        vals[i] = x; s += x; s2 += x * x;
    }
    __shared__ float rs[128], rs2[128];
    rs[tid] = s; rs2[tid] = s2;
    __syncthreads();
    for (int st = 64; st > 0; st >>= 1) {
        if (tid < st) { rs[tid] += rs[tid + st]; rs2[tid] += rs2[tid + st]; }
        __syncthreads();
    }
    const float mean = rs[0] / Dm;
    const float var  = rs2[0] / Dm - mean * mean;
    const float inv  = rsqrtf(var + 1e-5f);
    #pragma unroll
    for (int i = 0; i < 4; ++i) {
        int c = tid + i * 128;
        p[c] = (vals[i] - mean) * inv * g[c] + b[c];
    }
}

// ---------------------------------------------------------------------------
// Row softmax (small widths, one thread per row), scores pre-scaled by `scale`.
// ---------------------------------------------------------------------------
__global__ void softmax_rows(float* __restrict__ p, int rows, int width, float scale)
{
    int r = blockIdx.x * blockDim.x + threadIdx.x;
    if (r >= rows) return;
    float* row = p + (long)r * width;
    float mx = -1e30f;
    for (int i = 0; i < width; ++i) mx = fmaxf(mx, row[i] * scale);
    float s = 0.0f;
    for (int i = 0; i < width; ++i) {
        float e = __expf(row[i] * scale - mx);
        row[i] = e; s += e;
    }
    float inv = 1.0f / s;
    for (int i = 0; i < width; ++i) row[i] *= inv;
}

// ---------------------------------------------------------------------------
// Semantic gated residual:  mix = gate*retr + (1-gate)*x
// ---------------------------------------------------------------------------
__global__ void mix_kernel(const float* __restrict__ gate,
                           const float* __restrict__ retr,
                           const float* __restrict__ x,
                           float* __restrict__ mix, long n)
{
    long i = (long)blockIdx.x * blockDim.x + threadIdx.x;
    if (i < n) {
        float g = gate[i];
        mix[i] = g * retr[i] + (1.0f - g) * x[i];
    }
}

// ---------------------------------------------------------------------------
// Controller: logits = h1 @ C2w + C2b (3-wide), softmax, 3-way fuse -> out.
// One block (128 thr) per token.
// ---------------------------------------------------------------------------
__global__ __launch_bounds__(128) void fuse_kernel(
    const float* __restrict__ h1, const float* __restrict__ C2w,
    const float* __restrict__ C2b,
    const float* __restrict__ wout, const float* __restrict__ eout,
    const float* __restrict__ sout, float* __restrict__ out, int Dm)
{
    const int t   = blockIdx.x;
    const int tid = threadIdx.x;
    __shared__ float l0[128], l1[128], l2[128];
    __shared__ float w[3];

    float h = h1[(long)t * 128 + tid];
    l0[tid] = h * C2w[tid * 3 + 0];
    l1[tid] = h * C2w[tid * 3 + 1];
    l2[tid] = h * C2w[tid * 3 + 2];
    __syncthreads();
    for (int s = 64; s > 0; s >>= 1) {
        if (tid < s) { l0[tid] += l0[tid + s]; l1[tid] += l1[tid + s]; l2[tid] += l2[tid + s]; }
        __syncthreads();
    }
    if (tid == 0) {
        float a = l0[0] + C2b[0], b = l1[0] + C2b[1], c = l2[0] + C2b[2];
        float mx = fmaxf(a, fmaxf(b, c));
        float ea = __expf(a - mx), eb = __expf(b - mx), ec = __expf(c - mx);
        float inv = 1.0f / (ea + eb + ec);
        w[0] = ea * inv; w[1] = eb * inv; w[2] = ec * inv;
    }
    __syncthreads();
    #pragma unroll
    for (int i = 0; i < 4; ++i) {
        int c = tid + i * 128;
        long idx = (long)t * Dm + c;
        out[idx] = w[0] * wout[idx] + w[1] * eout[idx] + w[2] * sout[idx];
    }
}

// ---------------------------------------------------------------------------
// Host-side launcher
// ---------------------------------------------------------------------------
static void gemm(hipStream_t st, const float* A, const float* B, float* C,
                 const float* bias, int M_, int N_, int K_,
                 int lda, int ldb, int ldc,
                 int Z, int Hdiv,
                 long sAb, long sAh, long sBb, long sBh, long sCb, long sCh,
                 int transB, int act, float scale)
{
    dim3 grid((M_ + TM - 1) / TM, (N_ + TN - 1) / TN, Z);
    hipLaunchKernelGGL(gemm_wmma, grid, dim3(128), 0, st,
                       A, B, C, bias, M_, N_, K_, lda, ldb, ldc, Hdiv,
                       sAb, sAh, sBb, sBh, sCb, sCh, transB, act, scale);
}

extern "C" void kernel_launch(void* const* d_in, const int* in_sizes, int n_in,
                              void* d_out, int out_size, void* d_ws, size_t ws_size,
                              hipStream_t stream)
{
    // dims
    const int cB = 2, cL = 1024, cD = 512, cM = 64, cSE = 32, cSS = 64, cH = 4,
              cBOT = 128;
    const int N = cB * cL;            // 2048 tokens
    const int hd = cD / cH;           // 128

    // inputs (setup_inputs order)
    const float* x     = (const float*)d_in[0];
    const float* Wq    = (const float*)d_in[1];
    const float* Wk    = (const float*)d_in[2];
    const float* Wv    = (const float*)d_in[3];
    const float* Wf    = (const float*)d_in[4];
    const float* bf    = (const float*)d_in[5];
    const float* Wo    = (const float*)d_in[6];
    const float* ln_g  = (const float*)d_in[7];
    const float* ln_b  = (const float*)d_in[8];
    const float* eslots= (const float*)d_in[9];
    const float* Eq    = (const float*)d_in[10];
    const float* Ek    = (const float*)d_in[11];
    const float* Ev    = (const float*)d_in[12];
    const float* Eo    = (const float*)d_in[13];
    const float* know  = (const float*)d_in[14];
    const float* Sin   = (const float*)d_in[15];
    const float* Sg_w  = (const float*)d_in[16];
    const float* Sg_b  = (const float*)d_in[17];
    const float* So    = (const float*)d_in[18];
    const float* C1w   = (const float*)d_in[19];
    const float* C1b   = (const float*)d_in[20];
    const float* C2w   = (const float*)d_in[21];
    const float* C2b   = (const float*)d_in[22];
    float* out = (float*)d_out;

    // workspace layout (floats)
    float* ws = (float*)d_ws;
    long off = 0;
    float* q_     = ws + off; off += (long)N * cM;        // 2048x64
    float* k_     = ws + off; off += (long)N * cM;
    float* f_     = ws + off; off += (long)N * cM;
    float* v_     = ws + off; off += (long)N * cD;        // 2048x512
    float* y_     = ws + off; off += (long)N * cD;        // scan out / LN in-place
    float* wout   = ws + off; off += (long)N * cD;
    float* eq_    = ws + off; off += (long)N * cD;
    float* ek_    = ws + off; off += (long)cSE * cD;      // 32x512
    float* ev_    = ws + off; off += (long)cSE * cD;
    float* scores = ws + off; off += (long)cB * cH * cL * cSE; // 8192x32
    float* ehead  = ws + off; off += (long)N * cD;
    float* eout   = ws + off; off += (long)N * cD;
    float* qs_    = ws + off; off += (long)N * cD;
    float* sscr   = ws + off; off += (long)N * cSS;       // 2048x64
    float* retr   = ws + off; off += (long)N * cD;
    float* gate   = ws + off; off += (long)N * cD;
    float* mixb   = ws + off; off += (long)N * cD;
    float* sout   = ws + off; off += (long)N * cD;
    float* h1_    = ws + off; off += (long)N * cBOT;      // 2048x128
    (void)ws_size; (void)out_size; (void)in_sizes; (void)n_in;

    const float qscale = 0.125f;                    // M^-0.5
    const float escale = 1.0f / sqrtf((float)hd);   // 1/sqrt(128)
    const float sscale = 1.0f / sqrtf((float)cD);   // 1/sqrt(512)

    // ---- token-parallel projections (all WMMA GEMMs) ----
    gemm(stream, x, Wq, q_, nullptr, N, cM, cD, cD, cM, cM, 1,1, 0,0,0,0,0,0, 0, ACT_ELU1, qscale);
    gemm(stream, x, Wk, k_, nullptr, N, cM, cD, cD, cM, cM, 1,1, 0,0,0,0,0,0, 0, ACT_ELU1, 1.0f);
    gemm(stream, x, Wf, f_, bf,      N, cM, cD, cD, cM, cM, 1,1, 0,0,0,0,0,0, 0, ACT_SIGMOID, 1.0f);
    gemm(stream, x, Wv, v_, nullptr, N, cD, cD, cD, cD, cD, 1,1, 0,0,0,0,0,0, 0, ACT_NONE, 1.0f);
    gemm(stream, x, Eq, eq_, nullptr,N, cD, cD, cD, cD, cD, 1,1, 0,0,0,0,0,0, 0, ACT_NONE, 1.0f);
    gemm(stream, x, Sin, qs_, nullptr,N, cD, cD, cD, cD, cD, 1,1, 0,0,0,0,0,0, 0, ACT_NONE, 1.0f);
    gemm(stream, x, Sg_w, gate, Sg_b, N, cD, cD, cD, cD, cD, 1,1, 0,0,0,0,0,0, 0, ACT_SIGMOID, 1.0f);
    gemm(stream, x, C1w, h1_, C1b,   N, cBOT, cD, cD, cBOT, cBOT, 1,1, 0,0,0,0,0,0, 0, ACT_SILU, 1.0f);
    // slot projections (shared across batch; Mrows=32 -> guarded path)
    gemm(stream, eslots, Ek, ek_, nullptr, cSE, cD, cD, cD, cD, cD, 1,1, 0,0,0,0,0,0, 0, ACT_NONE, 1.0f);
    gemm(stream, eslots, Ev, ev_, nullptr, cSE, cD, cD, cD, cD, cD, 1,1, 0,0,0,0,0,0, 0, ACT_NONE, 1.0f);

    // ---- WorkingMemory scan + LN + output projection ----
    hipLaunchKernelGGL(wm_scan, dim3(cD / 64, cB), dim3(64), 0, stream,
                       q_, k_, f_, v_, y_, cL);
    hipLaunchKernelGGL(layernorm_ip, dim3(N), dim3(128), 0, stream,
                       y_, ln_g, ln_b, cD);
    gemm(stream, y_, Wo, wout, nullptr, N, cD, cD, cD, cD, cD, 1,1, 0,0,0,0,0,0, 0, ACT_NONE, 1.0f);

    // ---- EpisodicMemory: per-(b,h) scores = qh @ kh^T, softmax, attn @ vh ----
    gemm(stream, eq_, ek_, scores, nullptr,
         cL, cSE, hd, cD, cD, cSE,
         cB * cH, cH,
         (long)cL * cD, (long)hd,            // A offsets: b, h
         0, (long)hd,                        // B offsets: (shared), h
         (long)cH * cL * cSE, (long)cL * cSE,// C offsets: b, h
         1 /*transB*/, ACT_NONE, 1.0f);
    hipLaunchKernelGGL(softmax_rows, dim3((cB * cH * cL + 255) / 256), dim3(256), 0, stream,
                       scores, cB * cH * cL, cSE, escale);
    gemm(stream, scores, ev_, ehead, nullptr,
         cL, hd, cSE, cSE, cD, cD,
         cB * cH, cH,
         (long)cH * cL * cSE, (long)cL * cSE,
         0, (long)hd,
         (long)cL * cD, (long)hd,
         0, ACT_NONE, 1.0f);
    gemm(stream, ehead, Eo, eout, nullptr, N, cD, cD, cD, cD, cD, 1,1, 0,0,0,0,0,0, 0, ACT_NONE, 1.0f);

    // ---- SemanticMemory: scores = qs @ know^T, softmax, retr = attn @ know ----
    gemm(stream, qs_, know, sscr, nullptr, N, cSS, cD, cD, cD, cSS,
         1,1, 0,0,0,0,0,0, 1 /*transB*/, ACT_NONE, 1.0f);
    hipLaunchKernelGGL(softmax_rows, dim3((N + 255) / 256), dim3(256), 0, stream,
                       sscr, N, cSS, sscale);
    gemm(stream, sscr, know, retr, nullptr, N, cD, cSS, cSS, cD, cD,
         1,1, 0,0,0,0,0,0, 0, ACT_NONE, 1.0f);
    hipLaunchKernelGGL(mix_kernel, dim3((N * cD + 255) / 256), dim3(256), 0, stream,
                       gate, retr, x, mixb, (long)N * cD);
    gemm(stream, mixb, So, sout, nullptr, N, cD, cD, cD, cD, cD,
         1,1, 0,0,0,0,0,0, 0, ACT_NONE, 1.0f);

    // ---- Controller fusion -> d_out ----
    hipLaunchKernelGGL(fuse_kernel, dim3(N), dim3(128), 0, stream,
                       h1_, C2w, C2b, wout, eout, sout, out, cD);
}
